// sparse_block_67594195304881
// MI455X (gfx1250) — matrix operand order repaired and checked by reading, not compiled, not run on hardware
//
#include <hip/hip_runtime.h>
#include <hip/hip_bf16.h>

typedef __attribute__((ext_vector_type(16))) _Float16 v16h;
typedef __attribute__((ext_vector_type(8)))  _Float16 v8h;
typedef __attribute__((ext_vector_type(8)))  float    v8f;

#define HW     (512*512)
#define NCELL  1156        // 4 * 17 * 17
#define EPSBN  1e-5f

// ---------------- WMMA fragment helpers (layouts per CDNA5 ISA 7.12.2) ----------------
// A 16x32 f16: lane<16 -> row M=lane, halves[0:8]=K 0..7, halves[8:16]=K 16..23
//              lane>=16 -> row M=lane-16, halves = K 8..15 / 24..31   (sel = lane>>4)
template<int RS>
__device__ __forceinline__ v16h afrag(const _Float16* sIn, int px, int cb, int sel){
  const _Float16* r = sIn + px*RS + cb + sel*8;
  v8h lo = *(const v8h*)(r);
  v8h hi = *(const v8h*)(r + 16);
  return __builtin_shufflevector(lo, hi, 0,1,2,3,4,5,6,7,8,9,10,11,12,13,14,15);
}
// B 32x16 f16: lane<16 -> col N=lane, K 0..7 / 16..23 ; lane>=16 -> col N=lane-16, K 8..15 / 24..31
template<int RS>
__device__ __forceinline__ v16h bfragLds(const _Float16* sW, int co, int cb, int sel){
  const _Float16* r = sW + co*RS + cb + sel*8;
  v8h lo = *(const v8h*)(r);
  v8h hi = *(const v8h*)(r + 16);
  return __builtin_shufflevector(lo, hi, 0,1,2,3,4,5,6,7,8,9,10,11,12,13,14,15);
}
__device__ __forceinline__ v16h bfragGlb(const _Float16* w, int row, int cb, int sel){
  const _Float16* r = w + (size_t)row*128 + cb + sel*8;
  v8h lo = *(const v8h*)(r);
  v8h hi = *(const v8h*)(r + 16);
  return __builtin_shufflevector(lo, hi, 0,1,2,3,4,5,6,7,8,9,10,11,12,13,14,15);
}
__device__ __forceinline__ v8f wmma16(v16h a, v16h b, v8f c){
  return __builtin_amdgcn_wmma_f32_16x16x32_f16(false, a, false, b, (short)0, c, false, false);
}

// ---------------- K0: weights -> f16 (w2 rearranged to [tap][co][ci]) ----------------
__global__ void k0_prep(const float* __restrict__ ccw, const float* __restrict__ w1,
                        const float* __restrict__ w2,  const float* __restrict__ w3,
                        _Float16* __restrict__ ccwh, _Float16* __restrict__ w1h,
                        _Float16* __restrict__ w2t,  _Float16* __restrict__ w3h){
  int t = blockIdx.x*256 + threadIdx.x;
  if (t < 147456){
    int tap = t / 16384, rem = t % 16384, co = rem >> 7, ci = rem & 127;
    w2t[t] = (_Float16)w2[(size_t)(co*128 + ci)*9 + tap];
  } else if (t < 151552){ int i = t - 147456; ccwh[i] = (_Float16)ccw[i]; }
  else if (t < 159744){ int i = t - 151552; w1h[i] = (_Float16)w1[i]; }
  else if (t < 167936){ int i = t - 159744; w3h[i] = (_Float16)w3[i]; }
}

// ---------------- K_mask: pooled-mask flags + deterministic scan compaction ----------
// pooled cell (n,gh,gw) active  <=>  any overlapped aligned 32x32 mask block nonzero
__global__ void k_mask(const int* __restrict__ mask, int* __restrict__ list){
  __shared__ int sflag[256];
  __shared__ int sbase;
  const int t = threadIdx.x;
  if (t == 0) sbase = 0;
  __syncthreads();
  for (int chunk = 0; chunk < 5; ++chunk){
    int id = chunk*256 + t;
    int f = 0;
    if (id < NCELL){
      int n = id / 289, r = id % 289, gh = r / 17, gw = r % 17;
      for (int a = 0; a < 2; ++a){
        int br = gh - 1 + a; if (br < 0 || br >= 16) continue;
        for (int b = 0; b < 2; ++b){
          int bc = gw - 1 + b; if (bc < 0 || bc >= 16) continue;
          if (mask[(size_t)n*HW + (br*32)*512 + bc*32] != 0) f = 1;
        }
      }
    }
    sflag[t] = f;
    __syncthreads();
    for (int off = 1; off < 256; off <<= 1){
      int v = 0;
      if (t >= off) v = sflag[t-off];
      __syncthreads();
      sflag[t] += v;
      __syncthreads();
    }
    if (f) list[1 + sbase + sflag[t] - 1] = id;
    __syncthreads();
    if (t == 0) sbase += sflag[255];
    __syncthreads();
  }
  if (t == 0) list[0] = sbase;
}

// ---------------- K1: channel-control conv1x1 (64->64) + ReLU -> y f16 [px][64] -----
__global__ void k1_ccconv(const float* __restrict__ x, const _Float16* __restrict__ wcc,
                          const float* __restrict__ ccb, _Float16* __restrict__ y,
                          float* __restrict__ part){
  extern __shared__ char smem[];
  _Float16* sIn = (_Float16*)smem;        // [1024][72]
  _Float16* sW  = sIn + 1024*72;          // [64][72]
  const int t = threadIdx.x;
  const int p0 = blockIdx.x * 1024;
  const int n  = p0 >> 18;
  const int prem = p0 & (HW - 1);
  for (int k = t; k < 64*64; k += 256) sW[(k>>6)*72 + (k&63)] = wcc[k];
  for (int k = t; k < 16384; k += 256){
    int ci = k >> 8;
    int pxc = (k & 255) << 2;
    float4 v = *(const float4*)(x + (((size_t)(n*64 + ci)) << 18) + prem + pxc);
    sIn[(pxc+0)*72 + ci] = (_Float16)v.x;
    sIn[(pxc+1)*72 + ci] = (_Float16)v.y;
    sIn[(pxc+2)*72 + ci] = (_Float16)v.z;
    sIn[(pxc+3)*72 + ci] = (_Float16)v.w;
  }
  __syncthreads();
  const int wv = t >> 5, lane = t & 31, sel = lane >> 4, nn = lane & 15;
  const int co = (wv & 3)*16 + nn;
  const int mhalf = wv >> 2;
  const float bias = ccb[co];
  float s = 0.f, s2 = 0.f;
  for (int mg = 0; mg < 8; ++mg){
    const int mt0 = mhalf*32 + mg*4;
    v8f z = {};
    v8f acc[4]; acc[0]=z; acc[1]=z; acc[2]=z; acc[3]=z;
    #pragma unroll
    for (int cb = 0; cb < 64; cb += 32){
      v16h bf = bfragLds<72>(sW, co, cb, sel);
      #pragma unroll
      for (int i = 0; i < 4; ++i)
        acc[i] = wmma16(afrag<72>(sIn, (mt0+i)*16 + nn, cb, sel), bf, acc[i]);
    }
    #pragma unroll
    for (int i = 0; i < 4; ++i){
      const int pxb = (mt0+i)*16 + sel*8;
      #pragma unroll
      for (int q = 0; q < 8; ++q){
        float v = acc[i][q] + bias; v = v > 0.f ? v : 0.f;
        s += v; s2 += v*v;
        y[((size_t)(p0 + pxb + q))*64 + co] = (_Float16)v;
      }
    }
  }
  s  += __shfl_xor(s, 16, 32);
  s2 += __shfl_xor(s2, 16, 32);
  if (lane < 16){
    const size_t slot = (size_t)blockIdx.x*2 + mhalf;
    part[(slot*64 + co)*2]   = s;
    part[(slot*64 + co)*2+1] = s2;
  }
}

// ---------------- fixed-order BN stats reduction -> per-channel affine (a, b) -------
__global__ void k_reduce(const float* __restrict__ part, const float* __restrict__ gamma,
                         const float* __restrict__ beta, float* __restrict__ aff,
                         int C, int fixedSlots, int mult, const int* __restrict__ list,
                         float fixedN){
  int c = threadIdx.x; if (c >= C) return;
  int slots; float nelem;
  if (list){ int cnt = list[0]; slots = cnt*mult; nelem = (float)cnt * 1024.0f; }
  else     { slots = fixedSlots; nelem = fixedN; }
  double s = 0.0, s2 = 0.0;
  for (int i = 0; i < slots; ++i){
    s  += (double)part[((size_t)i*C + c)*2];
    s2 += (double)part[((size_t)i*C + c)*2 + 1];
  }
  float a = 0.f, b = 0.f;
  if (nelem > 0.f){
    float mean = (float)(s / (double)nelem);
    float var  = (float)(s2 / (double)nelem) - mean*mean;
    float inv  = rsqrtf(var + EPSBN);
    a = gamma[c] * inv;
    b = beta[c] - mean * a;
  }
  aff[2*c] = a; aff[2*c+1] = b;
}

// ---------------- zero-fill d_out (borders + base for scatter-add) ------------------
__global__ void k_zero(float4* __restrict__ o, long n4){
  for (long i = (long)blockIdx.x*blockDim.x + threadIdx.x; i < n4; i += (long)gridDim.x*blockDim.x)
    o[i] = make_float4(0.f, 0.f, 0.f, 0.f);
}

// ---------------- K3: out interior = affine_cc(y), NCHW via LDS transpose -----------
__global__ void k3_writebase(const _Float16* __restrict__ y, const float* __restrict__ aff,
                             float* __restrict__ out){
  extern __shared__ char smem[];
  _Float16* sT = (_Float16*)smem;   // [64][1032]
  const int t = threadIdx.x;
  const int p0 = blockIdx.x * 1024;
  for (int k = t; k < 8192; k += 256){
    int px = k >> 3, c8 = (k & 7) << 3;
    v8h v = *(const v8h*)(y + ((size_t)(p0 + px))*64 + c8);
    #pragma unroll
    for (int j = 0; j < 8; ++j) sT[(c8+j)*1032 + px] = v[j];
  }
  __syncthreads();
  const int n  = p0 >> 18;
  const int h0 = (p0 & (HW - 1)) >> 9;
  for (int k = t; k < 16384; k += 256){
    int c = k >> 8, rl = (k >> 7) & 1, xc = (k & 127) << 2;
    float a = aff[2*c], b = aff[2*c+1];
    const _Float16* r = sT + c*1032 + rl*512 + xc;
    float4 o;
    o.x = a*(float)r[0] + b; o.y = a*(float)r[1] + b;
    o.z = a*(float)r[2] + b; o.w = a*(float)r[3] + b;
    *(float4*)(out + (((size_t)(n*64 + c)*544) + 16 + h0 + rl)*544 + 16 + xc) = o;
  }
}

// ---------------- K5: gather(affine_cc, zero-pad) + conv1x1 64->128 + ReLU ----------
__global__ void k5_conv1(const _Float16* __restrict__ y, const float* __restrict__ affcc,
                         const int* __restrict__ list, const _Float16* __restrict__ w1h,
                         const float* __restrict__ b1, _Float16* __restrict__ g1,
                         float* __restrict__ part){
  const int cnt = list[0];
  const int bid = blockIdx.x; if (bid >= cnt) return;
  const int id = list[1 + bid];
  const int n = id/289, r0 = id%289, gh = r0/17, gw = r0%17;
  extern __shared__ char smem[];
  _Float16* sIn = (_Float16*)smem;          // [1024][72]
  _Float16* sW  = sIn + 1024*72;            // [128][72]
  float*    sA  = (float*)(sW + 128*72);    // 128 floats
  const int t = threadIdx.x;
  for (int k = t; k < 128; k += 256) sA[k] = affcc[k];
  for (int k = t; k < 128*64; k += 256) sW[(k>>6)*72 + (k&63)] = w1h[k];
  __syncthreads();
  const int y0 = gh*32 - 16, x0 = gw*32 - 16;
  for (int k = t; k < 8192; k += 256){
    int px = k >> 3, c8 = (k & 7) << 3;
    int yy = y0 + (px >> 5), xx = x0 + (px & 31);
    v8h ov;
    if ((unsigned)yy < 512u && (unsigned)xx < 512u){
      v8h v = *(const v8h*)(y + ((size_t)n*HW + yy*512 + xx)*64 + c8);
      #pragma unroll
      for (int j = 0; j < 8; ++j)
        ov[j] = (_Float16)(sA[2*(c8+j)]*(float)v[j] + sA[2*(c8+j)+1]);
    } else {
      #pragma unroll
      for (int j = 0; j < 8; ++j) ov[j] = (_Float16)0.f;
    }
    *(v8h*)(sIn + px*72 + c8) = ov;
  }
  __syncthreads();
  const int wv = t >> 5, lane = t & 31, sel = lane >> 4, nn = lane & 15;
  const int co = wv*16 + nn;
  const float bias = b1[co];
  float s = 0.f, s2 = 0.f;
  for (int mg = 0; mg < 16; ++mg){
    const int mt0 = mg*4;
    v8f z = {};
    v8f acc[4]; acc[0]=z; acc[1]=z; acc[2]=z; acc[3]=z;
    #pragma unroll
    for (int cb = 0; cb < 64; cb += 32){
      v16h bf = bfragLds<72>(sW, co, cb, sel);
      #pragma unroll
      for (int i = 0; i < 4; ++i)
        acc[i] = wmma16(afrag<72>(sIn, (mt0+i)*16 + nn, cb, sel), bf, acc[i]);
    }
    #pragma unroll
    for (int i = 0; i < 4; ++i){
      const int pxb = (mt0+i)*16 + sel*8;
      #pragma unroll
      for (int q = 0; q < 8; ++q){
        float v = acc[i][q] + bias; v = v > 0.f ? v : 0.f;
        s += v; s2 += v*v;
        g1[((size_t)bid*1024 + pxb + q)*128 + co] = (_Float16)v;
      }
    }
  }
  s  += __shfl_xor(s, 16, 32);
  s2 += __shfl_xor(s2, 16, 32);
  if (lane < 16){
    part[((size_t)bid*128 + co)*2]   = s;
    part[((size_t)bid*128 + co)*2+1] = s2;
  }
}

// ---------------- K7: conv3x3 128->128 (pad 1) + ReLU, implicit GEMM ---------------
__global__ void k7_conv2(const _Float16* __restrict__ g1, const float* __restrict__ aff1,
                         const int* __restrict__ list, const _Float16* __restrict__ w2t,
                         const float* __restrict__ b2, _Float16* __restrict__ g2,
                         float* __restrict__ part){
  const int cnt = list[0];
  const int bid = blockIdx.x; if (bid >= cnt) return;
  extern __shared__ char smem[];
  _Float16* sIn = (_Float16*)smem;            // [1024][136]
  float*    sA  = (float*)(sIn + 1024*136);   // 256 floats
  const int t = threadIdx.x;
  for (int k = t; k < 256; k += 256) sA[k] = aff1[k];
  __syncthreads();
  for (int k = t; k < 16384; k += 256){
    int px = k >> 4, c8 = (k & 15) << 3;
    v8h v = *(const v8h*)(g1 + ((size_t)bid*1024 + px)*128 + c8);
    v8h ov;
    #pragma unroll
    for (int j = 0; j < 8; ++j)
      ov[j] = (_Float16)(sA[2*(c8+j)]*(float)v[j] + sA[2*(c8+j)+1]);
    *(v8h*)(sIn + px*136 + c8) = ov;
  }
  __syncthreads();
  const int wv = t >> 5, lane = t & 31, sel = lane >> 4, nn = lane & 15;
  const int co = wv*16 + nn;
  const float bias = b2[co];
  float s = 0.f, s2 = 0.f;
  for (int mg = 0; mg < 16; ++mg){
    const int mt0 = mg*4;
    v8f z = {};
    v8f acc[4]; acc[0]=z; acc[1]=z; acc[2]=z; acc[3]=z;
    int pxs[4];
    #pragma unroll
    for (int i = 0; i < 4; ++i) pxs[i] = (mt0+i)*16 + nn;
    for (int tap = 0; tap < 9; ++tap){
      const int dy = tap/3 - 1, dx = tap%3 - 1;
      int px2[4]; bool ok[4];
      #pragma unroll
      for (int i = 0; i < 4; ++i){
        int yy = (pxs[i] >> 5) + dy, xx = (pxs[i] & 31) + dx;
        ok[i]  = ((unsigned)yy < 32u) && ((unsigned)xx < 32u);
        px2[i] = ok[i] ? (yy*32 + xx) : pxs[i];
      }
      #pragma unroll
      for (int cb = 0; cb < 128; cb += 32){
        v16h bf = bfragGlb(w2t, tap*128 + co, cb, sel);
        #pragma unroll
        for (int i = 0; i < 4; ++i){
          v16h af = {};
          if (ok[i]) af = afrag<136>(sIn, px2[i], cb, sel);
          acc[i] = wmma16(af, bf, acc[i]);
        }
      }
    }
    #pragma unroll
    for (int i = 0; i < 4; ++i){
      const int pxb = (mt0+i)*16 + sel*8;
      #pragma unroll
      for (int q = 0; q < 8; ++q){
        float v = acc[i][q] + bias; v = v > 0.f ? v : 0.f;
        s += v; s2 += v*v;
        g2[((size_t)bid*1024 + pxb + q)*128 + co] = (_Float16)v;
      }
    }
  }
  s  += __shfl_xor(s, 16, 32);
  s2 += __shfl_xor(s2, 16, 32);
  if (lane < 16){
    part[((size_t)bid*128 + co)*2]   = s;
    part[((size_t)bid*128 + co)*2+1] = s2;
  }
}

// ---------------- K9: conv1x1 128->64 + ReLU ---------------------------------------
__global__ void k9_conv3(const _Float16* __restrict__ g2, const float* __restrict__ aff2,
                         const int* __restrict__ list, const _Float16* __restrict__ w3h,
                         const float* __restrict__ b3, _Float16* __restrict__ g3,
                         float* __restrict__ part){
  const int cnt = list[0];
  const int bid = blockIdx.x; if (bid >= cnt) return;
  extern __shared__ char smem[];
  _Float16* sIn = (_Float16*)smem;            // [1024][136]
  _Float16* sW  = sIn + 1024*136;             // [64][136]
  float*    sA  = (float*)(sW + 64*136);      // 256 floats
  const int t = threadIdx.x;
  for (int k = t; k < 256; k += 256) sA[k] = aff2[k];
  for (int k = t; k < 64*128; k += 256) sW[(k>>7)*136 + (k&127)] = w3h[k];
  __syncthreads();
  for (int k = t; k < 16384; k += 256){
    int px = k >> 4, c8 = (k & 15) << 3;
    v8h v = *(const v8h*)(g2 + ((size_t)bid*1024 + px)*128 + c8);
    v8h ov;
    #pragma unroll
    for (int j = 0; j < 8; ++j)
      ov[j] = (_Float16)(sA[2*(c8+j)]*(float)v[j] + sA[2*(c8+j)+1]);
    *(v8h*)(sIn + px*136 + c8) = ov;
  }
  __syncthreads();
  const int wv = t >> 5, lane = t & 31, sel = lane >> 4, nn = lane & 15;
  const int co = (wv & 3)*16 + nn;
  const int mhalf = wv >> 2;
  const float bias = b3[co];
  float s = 0.f, s2 = 0.f;
  for (int mg = 0; mg < 8; ++mg){
    const int mt0 = mhalf*32 + mg*4;
    v8f z = {};
    v8f acc[4]; acc[0]=z; acc[1]=z; acc[2]=z; acc[3]=z;
    #pragma unroll
    for (int cb = 0; cb < 128; cb += 32){
      v16h bf = bfragLds<136>(sW, co, cb, sel);
      #pragma unroll
      for (int i = 0; i < 4; ++i)
        acc[i] = wmma16(afrag<136>(sIn, (mt0+i)*16 + nn, cb, sel), bf, acc[i]);
    }
    #pragma unroll
    for (int i = 0; i < 4; ++i){
      const int pxb = (mt0+i)*16 + sel*8;
      #pragma unroll
      for (int q = 0; q < 8; ++q){
        float v = acc[i][q] + bias; v = v > 0.f ? v : 0.f;
        s += v; s2 += v*v;
        g3[((size_t)bid*1024 + pxb + q)*64 + co] = (_Float16)v;
      }
    }
  }
  s  += __shfl_xor(s, 16, 32);
  s2 += __shfl_xor(s2, 16, 32);
  if (lane < 16){
    const size_t slot = (size_t)bid*2 + mhalf;
    part[(slot*64 + co)*2]   = s;
    part[(slot*64 + co)*2+1] = s2;
  }
}

// ---------------- K8: scatter-add affine3(g3) into out -----------------------------
__global__ void k8_scatter(const _Float16* __restrict__ g3, const float* __restrict__ aff3,
                           const int* __restrict__ list, float* __restrict__ out){
  const int cnt = list[0];
  const int bid = blockIdx.x; if (bid >= cnt) return;
  const int id = list[1 + bid];
  const int n = id/289, r0 = id%289, gh = r0/17, gw = r0%17;
  extern __shared__ char smem[];
  _Float16* sT = (_Float16*)smem;   // [64][1032]
  const int t = threadIdx.x;
  for (int k = t; k < 8192; k += 256){
    int px = k >> 3, c8 = (k & 7) << 3;
    v8h v = *(const v8h*)(g3 + ((size_t)bid*1024 + px)*64 + c8);
    #pragma unroll
    for (int j = 0; j < 8; ++j) sT[(c8+j)*1032 + px] = v[j];
  }
  __syncthreads();
  for (int k = t; k < 16384; k += 256){
    int c = k >> 8, yy = (k >> 3) & 31, xc = (k & 7) << 2;
    float a = aff3[2*c], b = aff3[2*c+1];
    const _Float16* r = sT + c*1032 + yy*32 + xc;
    float4* p = (float4*)(out + (((size_t)(n*64 + c)*544) + gh*32 + yy)*544 + gw*32 + xc);
    float4 o = *p;
    o.x += a*(float)r[0] + b; o.y += a*(float)r[1] + b;
    o.z += a*(float)r[2] + b; o.w += a*(float)r[3] + b;
    *p = o;
  }
}

// ---------------- host launcher ----------------------------------------------------
extern "C" void kernel_launch(void* const* d_in, const int* in_sizes, int n_in,
                              void* d_out, int out_size, void* d_ws, size_t ws_size,
                              hipStream_t stream){
  (void)in_sizes; (void)n_in; (void)ws_size;
  const float* x    = (const float*)d_in[0];
  const int*   mask = (const int*)d_in[1];
  const float* cc_w = (const float*)d_in[2];
  const float* cc_b = (const float*)d_in[3];
  const float* cc_g = (const float*)d_in[4];
  const float* cc_be= (const float*)d_in[5];
  const float* w1   = (const float*)d_in[6];
  const float* b1   = (const float*)d_in[7];
  const float* g1g  = (const float*)d_in[8];
  const float* be1  = (const float*)d_in[9];
  const float* w2   = (const float*)d_in[10];
  const float* b2   = (const float*)d_in[11];
  const float* g2g  = (const float*)d_in[12];
  const float* be2  = (const float*)d_in[13];
  const float* w3   = (const float*)d_in[14];
  const float* b3   = (const float*)d_in[15];
  const float* g3g  = (const float*)d_in[16];
  const float* be3  = (const float*)d_in[17];
  float* out = (float*)d_out;

  char* ws = (char*)d_ws;
  size_t off = 0;
  auto take = [&](size_t bytes)->char*{
    char* p = ws + off; off = (off + bytes + 255) & ~(size_t)255; return p;
  };
  // y (134MB) aliased with g2 (303MB): y dead after K5, g2 written in K7.
  _Float16* yb   = (_Float16*)take(303038464);
  _Float16* g2b  = yb;
  _Float16* g1b  = (_Float16*)take(303038464);
  _Float16* g3b  = (_Float16*)take(151519232);
  _Float16* w2t  = (_Float16*)take(294912);
  _Float16* ccwh = (_Float16*)take(8192);
  _Float16* w1h  = (_Float16*)take(16384);
  _Float16* w3h  = (_Float16*)take(16384);
  float* partcc  = (float*)take(1048576);   // 2048 slots x 64 ch x 2
  float* part1   = (float*)take(1183744);   // 1156 x 128 x 2
  float* part2   = (float*)take(1183744);
  float* part3   = (float*)take(1183744);   // 2312 x 64 x 2
  float* affcc   = (float*)take(512);
  float* aff1    = (float*)take(1024);
  float* aff2    = (float*)take(1024);
  float* aff3    = (float*)take(512);
  int*   list    = (int*)take(4*(1 + NCELL));

  k0_prep<<<656, 256, 0, stream>>>(cc_w, w1, w2, w3, ccwh, w1h, w2t, w3h);
  k_mask<<<1, 256, 0, stream>>>(mask, list);
  k1_ccconv<<<1024, 256, 156672, stream>>>(x, ccwh, cc_b, yb, partcc);
  k_reduce<<<1, 128, 0, stream>>>(partcc, cc_g, cc_be, affcc, 64, 2048, 0, nullptr, 1048576.0f);
  k_zero<<<2048, 256, 0, stream>>>((float4*)out, (long)out_size/4);
  k3_writebase<<<1024, 256, 132096, stream>>>(yb, affcc, out);
  k5_conv1<<<NCELL, 256, 166400, stream>>>(yb, affcc, list, w1h, b1, g1b, part1);
  k_reduce<<<1, 128, 0, stream>>>(part1, g1g, be1, aff1, 128, 0, 1, list, 0.f);
  k7_conv2<<<NCELL, 256, 279552, stream>>>(g1b, aff1, list, w2t, b2, g2b, part2);
  k_reduce<<<1, 128, 0, stream>>>(part2, g2g, be2, aff2, 128, 0, 1, list, 0.f);
  k9_conv3<<<NCELL, 256, 296960, stream>>>(g2b, aff2, list, w3h, b3, g3b, part3);
  k_reduce<<<1, 128, 0, stream>>>(part3, g3g, be3, aff3, 64, 0, 2, list, 0.f);
  k8_scatter<<<NCELL, 256, 132096, stream>>>(g3b, aff3, list, out);
}
